// MultiHeadAttention_29721173689037
// MI455X (gfx1250) — compile-verified, matrix operand order checked by reading, not run on hardware
//
#include <hip/hip_runtime.h>
#include <stdint.h>

// ---------- CDNA5 WMMA / TDM types ----------
typedef __attribute__((ext_vector_type(16))) __bf16       v16bf;
typedef __attribute__((ext_vector_type(8)))  float        v8f;
typedef __attribute__((ext_vector_type(4)))  uint16_t     u16x4;
typedef __attribute__((ext_vector_type(8)))  uint16_t     u16x8;
typedef __attribute__((ext_vector_type(16))) uint16_t     u16x16;
typedef __attribute__((ext_vector_type(4)))  unsigned int u32x4;
typedef __attribute__((ext_vector_type(8)))  int          i32x8;
typedef __attribute__((ext_vector_type(4)))  int          i32x4;

__device__ __forceinline__ uint16_t f2bf(float f) {
  union { float f; uint32_t u; } c; c.f = f;
  uint32_t u = c.u + 0x7FFFu + ((c.u >> 16) & 1u);   // RNE
  return (uint16_t)(u >> 16);
}

// 16x32 A-fragment / 32x16 B-fragment per-lane load (bf16).
// lane holds K = kbase + hi*8 + [0..7] and kbase + 16 + hi*8 + [0..7] of its row.
__device__ __forceinline__ v16bf load_frag(const uint16_t* rowptr, int kbase, int hi) {
  u16x8 lo = *(const u16x8*)(rowptr + kbase + hi * 8);
  u16x8 hv = *(const u16x8*)(rowptr + kbase + 16 + hi * 8);
  u16x16 u;
#pragma unroll
  for (int i = 0; i < 8; ++i) { u[i] = lo[i]; u[8 + i] = hv[i]; }
  return __builtin_bit_cast(v16bf, u);
}

__device__ __forceinline__ v8f wmma_bf16(v16bf a, v16bf b, v8f c) {
  return __builtin_amdgcn_wmma_f32_16x16x32_bf16(false, a, false, b, (short)0, c,
                                                 false, false);
}

// ---------- Tensor Data Mover: 2D tile (bf16) global -> LDS ----------
// Descriptor packing per CDNA5 ISA ch.8: group0 = {flags, lds_addr, global_addr,
// type=2}; group1 = {mask/data_size, tensor dims, tile dims, dim0 stride}.
__device__ __forceinline__ void tdm_load_2d(uint32_t lds_byte_off, const void* gaddr,
                                            uint32_t tensor_d0, uint32_t tensor_d1,
                                            uint64_t stride0_elems,
                                            uint32_t tile_d0, uint32_t tile_d1) {
  const uint64_t ga = (uint64_t)gaddr;
  u32x4 g0;
  g0[0] = 1u;                                           // count=1, no gather
  g0[1] = lds_byte_off;                                 // LDS dest (bytes)
  g0[2] = (uint32_t)(ga & 0xFFFFFFFFu);                 // global_addr[31:0]
  g0[3] = (uint32_t)((ga >> 32) & 0x1FFFFFFu) | (2u << 30);  // addr[56:32]|type=2
  i32x8 g1;
  g1[0] = (int)(1u << 16);                              // wg_mask=0, data_size=2B
  g1[1] = (int)((tensor_d0 & 0xFFFFu) << 16);           // tensor_dim0[15:0]
  g1[2] = (int)((tensor_d0 >> 16) | ((tensor_d1 & 0xFFFFu) << 16));
  g1[3] = (int)((tensor_d1 >> 16) | (tile_d0 << 16));   // dim1 hi | tile_dim0
  g1[4] = (int)tile_d1;                                 // tile_dim1, tile_dim2=0
  g1[5] = (int)(uint32_t)(stride0_elems & 0xFFFFFFFFu); // dim0 stride
  g1[6] = (int)(uint32_t)((stride0_elems >> 32) & 0xFFFFu);
  g1[7] = 0;
  i32x4 z4 = {0, 0, 0, 0};
#if __clang_major__ >= 23
  i32x8 z8 = {0, 0, 0, 0, 0, 0, 0, 0};
  __builtin_amdgcn_tensor_load_to_lds(g0, g1, z4, z4, z8, 0);
#else
  __builtin_amdgcn_tensor_load_to_lds(g0, g1, z4, z4, 0);
#endif
}

// ---------- fp32 -> bf16 conversion ----------
__global__ void f2bf_kernel(const float* __restrict__ src,
                            uint16_t* __restrict__ dst, int n) {
  int i4 = (blockIdx.x * blockDim.x + threadIdx.x) * 4;
  if (i4 < n) {
    const float4 v = *(const float4*)(src + i4);
    u16x4 o;
    o[0] = f2bf(v.x); o[1] = f2bf(v.y); o[2] = f2bf(v.z); o[3] = f2bf(v.w);
    *(u16x4*)(dst + i4) = o;
  }
}

// ---------- GEMM: C[M,N] = A[M,K] * B[N,K]^T, TDM-fed double buffer ----------
// Block: 256 threads (8 waves). Block tile 128x128, K-step 32.
// Waves arranged 4(m) x 2(n); each wave computes 32x64 (8 WMMA / K-step).
// Wave 0 drives the Tensor Data Mover; s_wait_tensorcnt gates the ping-pong.
template <bool OUT_F32>
__global__ __launch_bounds__(256)
void gemm_bt_kernel(const uint16_t* __restrict__ A, const uint16_t* __restrict__ Bw,
                    const float* __restrict__ bias,
                    float* __restrict__ Cf, uint16_t* __restrict__ Cb,
                    int M, int N, int K) {
  constexpr int PAN = 128 * 32;                 // one panel, elements
  __shared__ __align__(16) uint16_t smem[2 * 2 * PAN];   // 32 KB, 2 bufs x (A,B)
  const int tid  = threadIdx.x;
  const int wave = tid >> 5, lane = tid & 31;
  const int lm = lane & 15, hi = lane >> 4;
  const int wm = wave >> 1, wn = wave & 1;
  const int m0 = blockIdx.y * 128, n0 = blockIdx.x * 128;

  v8f acc[2][4] = {};

  const int niter = K / 32;
  if (wave == 0) {   // prologue DMA into buffer 0
    tdm_load_2d(0,           A  + (size_t)m0 * K, (uint32_t)K, (uint32_t)M,
                (uint64_t)K, 32, 128);
    tdm_load_2d(PAN * 2,     Bw + (size_t)n0 * K, (uint32_t)K, (uint32_t)N,
                (uint64_t)K, 32, 128);
  }

  for (int i = 0; i < niter; ++i) {
    const int p  = i & 1;
    if (wave == 0) {
      if (i + 1 < niter) {   // DMA next panel into the other buffer
        const int k1 = (i + 1) * 32;
        const uint32_t ob = (uint32_t)((p ^ 1) * 2 * PAN * 2);
        tdm_load_2d(ob,           A  + (size_t)m0 * K + k1, (uint32_t)K,
                    (uint32_t)M, (uint64_t)K, 32, 128);
        tdm_load_2d(ob + PAN * 2, Bw + (size_t)n0 * K + k1, (uint32_t)K,
                    (uint32_t)N, (uint64_t)K, 32, 128);
        __builtin_amdgcn_s_wait_tensorcnt((short)2);   // current buffer landed
      } else {
        __builtin_amdgcn_s_wait_tensorcnt((short)0);
      }
    }
    __syncthreads();                                   // buffer p ready for all

    const uint16_t* pA = smem + p * 2 * PAN;
    const uint16_t* pB = pA + PAN;
    v16bf af0 = load_frag(&pA[(wm * 32 + lm) * 32], 0, hi);
    v16bf af1 = load_frag(&pA[(wm * 32 + 16 + lm) * 32], 0, hi);
#pragma unroll
    for (int j = 0; j < 4; ++j) {
      v16bf bf = load_frag(&pB[(wn * 64 + j * 16 + lm) * 32], 0, hi);
      acc[0][j] = wmma_bf16(af0, bf, acc[0][j]);
      acc[1][j] = wmma_bf16(af1, bf, acc[1][j]);
    }
    __syncthreads();               // done reading p before it is overwritten
  }

#pragma unroll
  for (int t = 0; t < 2; ++t)
#pragma unroll
    for (int j = 0; j < 4; ++j)
#pragma unroll
      for (int e = 0; e < 8; ++e) {
        const int row = m0 + wm * 32 + t * 16 + e + 8 * hi;
        const int col = n0 + wn * 64 + j * 16 + lm;
        const float v = acc[t][j][e];
        if (OUT_F32) Cf[(size_t)row * N + col] = v + bias[col];
        else         Cb[(size_t)row * N + col] = f2bf(v);
      }
}

// ---------- Causal flash attention ----------
// Grid: (S/64, H, B). Block: 128 threads (4 waves); wave w owns 16 query rows.
// K tile staged via TDM; V tile transposed through LDS for PV B-fragments.
__global__ __launch_bounds__(128)
void flash_attn_kernel(const uint16_t* __restrict__ Qb, const uint16_t* __restrict__ Kb,
                       const uint16_t* __restrict__ Vb, uint16_t* __restrict__ Ctx) {
  constexpr int S = 2048, D = 1024, HD = 64;
  constexpr int KLS = 64 * 64;          // K tile   [key][hd]
  constexpr int VTS = 64 * 72;          // V^T tile [hd][key] (padded)
  __shared__ __align__(16) uint16_t smem[KLS + VTS + 4 * 16 * 64];
  uint16_t* Kls = smem;
  uint16_t* Vt  = smem + KLS;
  uint16_t* Pls = smem + KLS + VTS;

  const int tid = threadIdx.x, wave = tid >> 5, lane = tid & 31;
  const int lm = lane & 15, hi = lane >> 4;
  const int qt = blockIdx.x, h = blockIdx.y, b = blockIdx.z;
  const size_t bs = (size_t)b * S;

  const uint16_t* qptr = Qb + (bs + qt * 64 + wave * 16 + lm) * D + h * HD;
  v16bf aQ[2];
  aQ[0] = load_frag(qptr, 0, hi);
  aQ[1] = load_frag(qptr, 32, hi);

  float mi[8], li[8];
  v8f O[4] = {};
#pragma unroll
  for (int e = 0; e < 8; ++e) { mi[e] = -1e30f; li[e] = 0.f; }
  const float sc = 0.125f * 1.44269504088896f;   // 1/sqrt(64) * log2(e)

  for (int kt = 0; kt <= qt; ++kt) {
    __syncthreads();                   // previous tile fully consumed
    if (wave == 0)                     // TDM: 64x64 K tile -> LDS
      tdm_load_2d(0, Kb + (bs + kt * 64) * D + h * HD,
                  (uint32_t)D, (uint32_t)(4 * S), (uint64_t)D, 64, 64);
    {  // cooperative transpose-stage of the V tile (overlaps the TDM)
      const int r  = tid >> 1;
      const int cb = (tid & 1) * 32;
      const uint16_t* vrow = Vb + (bs + kt * 64 + r) * D + h * HD + cb;
#pragma unroll
      for (int c = 0; c < 32; ++c) Vt[(cb + c) * 72 + r] = vrow[c];
    }
    if (wave == 0) __builtin_amdgcn_s_wait_tensorcnt((short)0);
    __syncthreads();

    // S = Q K^T (4 n-subtiles x 2 K-chunks = 8 WMMA), K frags from LDS
    v8f s[4];
#pragma unroll
    for (int j = 0; j < 4; ++j) {
      const uint16_t* kp = &Kls[(j * 16 + lm) * 64];
      v8f t = {};
      t = wmma_bf16(aQ[0], load_frag(kp, 0, hi), t);
      t = wmma_bf16(aQ[1], load_frag(kp, 32, hi), t);
      s[j] = t;
    }

    const bool diag = (kt == qt);
#pragma unroll
    for (int j = 0; j < 4; ++j)
#pragma unroll
      for (int e = 0; e < 8; ++e) {
        float v = s[j][e] * sc;
        if (diag) {
          const int qg = wave * 16 + e + 8 * hi;
          const int kg = j * 16 + lm;
          if (kg > qg) v = -1e30f;
        }
        s[j][e] = v;
      }

    // online softmax (row = one 16-lane half at one accumulator index)
    uint16_t* Pw = &Pls[wave * 1024];
#pragma unroll
    for (int e = 0; e < 8; ++e) {
      float mx = fmaxf(fmaxf(s[0][e], s[1][e]), fmaxf(s[2][e], s[3][e]));
#pragma unroll
      for (int off = 8; off >= 1; off >>= 1)
        mx = fmaxf(mx, __shfl_xor(mx, off, 32));
      const float nm = fmaxf(mi[e], mx);
      const float alpha = __builtin_amdgcn_exp2f(mi[e] - nm);
      mi[e] = nm;
      float ls = 0.f;
      const int m = e + 8 * hi;
#pragma unroll
      for (int j = 0; j < 4; ++j) {
        const float p = __builtin_amdgcn_exp2f(s[j][e] - nm);
        ls += p;
        Pw[m * 64 + j * 16 + lm] = f2bf(p);   // C-layout -> LDS row-major
      }
#pragma unroll
      for (int off = 8; off >= 1; off >>= 1)
        ls += __shfl_xor(ls, off, 32);
      li[e] = li[e] * alpha + ls;
#pragma unroll
      for (int j = 0; j < 4; ++j) O[j][e] *= alpha;
    }

    asm volatile("s_wait_dscnt 0" ::: "memory");   // wave-local P RAW in LDS

    // O += P * V
#pragma unroll
    for (int c = 0; c < 2; ++c) {
      v16bf aP = load_frag(&Pw[lm * 64], c * 32, hi);
#pragma unroll
      for (int j = 0; j < 4; ++j) {
        v16bf bV = load_frag(&Vt[(j * 16 + lm) * 72], c * 32, hi);
        O[j] = wmma_bf16(aP, bV, O[j]);
      }
    }
  }

#pragma unroll
  for (int e = 0; e < 8; ++e) {
    const float inv = 1.0f / li[e];
    const int row = qt * 64 + wave * 16 + e + 8 * hi;
    uint16_t* orow = Ctx + (bs + row) * D + h * HD;
#pragma unroll
    for (int j = 0; j < 4; ++j) orow[j * 16 + lm] = f2bf(O[j][e] * inv);
  }
}

// ---------- host-side orchestration ----------
extern "C" void kernel_launch(void* const* d_in, const int* in_sizes, int n_in,
                              void* d_out, int out_size, void* d_ws, size_t ws_size,
                              hipStream_t stream) {
  const float* x  = (const float*)d_in[0];
  const float* Wq = (const float*)d_in[1];
  const float* Wk = (const float*)d_in[2];
  const float* Wv = (const float*)d_in[3];
  const float* Wo = (const float*)d_in[4];
  const float* bo = (const float*)d_in[5];
  float* out = (float*)d_out;

  const int Bn = 4, S = 2048, D = 1024, H = 16;
  const size_t MS = (size_t)Bn * S;   // 8192 rows

  uint16_t* ws  = (uint16_t*)d_ws;
  uint16_t* Xb  = ws;                       // MS*D
  uint16_t* Wqb = Xb  + MS * D;             // D*D
  uint16_t* Wkb = Wqb + (size_t)D * D;
  uint16_t* Wvb = Wkb + (size_t)D * D;
  uint16_t* Wob = Wvb + (size_t)D * D;
  uint16_t* Qb  = Wob + (size_t)D * D;      // MS*D
  uint16_t* Kb  = Qb  + MS * D;
  uint16_t* Vb  = Kb  + MS * D;
  uint16_t* Cb  = Vb  + MS * D;             // MS*D (context, bf16)

  const int cth = 256;
  auto cvt = [&](const float* s, uint16_t* d, size_t n) {
    const int blocks = (int)((n / 4 + cth - 1) / cth);
    f2bf_kernel<<<dim3(blocks), dim3(cth), 0, stream>>>(s, d, (int)n);
  };
  cvt(x,  Xb,  MS * D);
  cvt(Wq, Wqb, (size_t)D * D);
  cvt(Wk, Wkb, (size_t)D * D);
  cvt(Wv, Wvb, (size_t)D * D);
  cvt(Wo, Wob, (size_t)D * D);

  const dim3 gp(D / 128, (unsigned)(MS / 128));  // (8, 64)
  gemm_bt_kernel<false><<<gp, 256, 0, stream>>>(Xb, Wqb, nullptr, nullptr, Qb,
                                                (int)MS, D, D);
  gemm_bt_kernel<false><<<gp, 256, 0, stream>>>(Xb, Wkb, nullptr, nullptr, Kb,
                                                (int)MS, D, D);
  gemm_bt_kernel<false><<<gp, 256, 0, stream>>>(Xb, Wvb, nullptr, nullptr, Vb,
                                                (int)MS, D, D);

  flash_attn_kernel<<<dim3(S / 64, H, Bn), 128, 0, stream>>>(Qb, Kb, Vb, Cb);

  gemm_bt_kernel<true><<<gp, 256, 0, stream>>>(Cb, Wob, bo, out, nullptr,
                                               (int)MS, D, D);
}
